// MemNetE2E_66941360275674
// MI455X (gfx1250) — matrix operand order repaired and checked by reading, not compiled, not run on hardware
//
#include <hip/hip_runtime.h>
#include <hip/hip_bf16.h>
#include <math.h>

// ---------------------------------------------------------------------------
// MemNetE2E: cosine-sim top-5 retrieval on gfx1250 (wave32, WMMA bf16)
//  k1: row norms (+ pre-normalized bf16 key cache in ws, + ages baseline copy)
//  k2: WMMA sim + streaming per-chunk top-5 (32 queries x 4096 keys per WG;
//      A fragments register-resident in the PRE path)
//  k3: candidate merge, gather-means, age scatter
// ---------------------------------------------------------------------------

typedef __bf16  v8bf  __attribute__((ext_vector_type(8)));
typedef __bf16  v16bf __attribute__((ext_vector_type(16)));
typedef float   v8f   __attribute__((ext_vector_type(8)));

#define BQ      1024          // queries
#define SK      65536         // keys / slots
#define DD      256           // key dim
#define VV      256           // value dim
#define TOPK    5
#define CHUNK   4096          // keys per S-chunk (per blockIdx.y)
#define NCHUNK  (SK / CHUNK)  // 16
#define MWG     32            // queries per workgroup (2 WMMA row-tiles)
#define NQG     (BQ / MWG)    // 32 query groups
#define TPW     (CHUNK / 16 / 8) // 32 key-tiles per wave
#define KSTEPS  (DD / 32)     // 8 WMMA K-steps
#define NEG_INF (-3.402823466e+38f)
#define EPS_    (1e-8f)

struct Top5 {
    float v0, v1, v2, v3, v4;
    int   i0, i1, i2, i3, i4;
    __device__ __forceinline__ void init() {
        v0 = v1 = v2 = v3 = v4 = NEG_INF;
        i0 = i1 = i2 = i3 = i4 = 0;
    }
    __device__ __forceinline__ void insert(float s, int n) {
        if (s <= v4) return;
        v4 = s; i4 = n;
        if (v4 > v3) { float t=v3; v3=v4; v4=t; int u=i3; i3=i4; i4=u; }
        if (v3 > v2) { float t=v2; v2=v3; v3=t; int u=i2; i2=i3; i3=u; }
        if (v2 > v1) { float t=v1; v1=v2; v2=t; int u=i1; i1=i2; i2=u; }
        if (v1 > v0) { float t=v0; v0=v1; v1=t; int u=i0; i0=i1; i1=u; }
    }
};

// ---------------------------------------------------------------------------
// Kernel 1: one wave per row. Computes 1/max(||row||,eps); optionally emits
// the row pre-normalized in bf16 (key cache) and copies ages -> out_ages.
// ---------------------------------------------------------------------------
__global__ __launch_bounds__(256) void rownorm_kernel(
    const float* __restrict__ src, int rows, float* __restrict__ rcp_out,
    __bf16* __restrict__ bf_out,
    const float* __restrict__ ages_in, float* __restrict__ ages_out)
{
    const int wid  = threadIdx.x >> 5;
    const int lane = threadIdx.x & 31;
    const int row  = blockIdx.x * 8 + wid;
    if (row >= rows) return;

    const float4* p = (const float4*)(src + (size_t)row * DD + lane * 8);
    const float4 a = p[0], b = p[1];
    float s = a.x*a.x + a.y*a.y + a.z*a.z + a.w*a.w
            + b.x*b.x + b.y*b.y + b.z*b.z + b.w*b.w;
    s += __shfl_xor(s, 16);
    s += __shfl_xor(s, 8);
    s += __shfl_xor(s, 4);
    s += __shfl_xor(s, 2);
    s += __shfl_xor(s, 1);                      // butterfly: all lanes hold sum
    const float rcp = 1.0f / fmaxf(sqrtf(s), EPS_);
    if (lane == 0) {
        rcp_out[row] = rcp;
        if (ages_out) ages_out[row] = ages_in[row];
    }
    if (bf_out) {                               // pre-normalized bf16 row cache
        v8bf o;
        o[0]=(__bf16)(a.x*rcp); o[1]=(__bf16)(a.y*rcp);
        o[2]=(__bf16)(a.z*rcp); o[3]=(__bf16)(a.w*rcp);
        o[4]=(__bf16)(b.x*rcp); o[5]=(__bf16)(b.y*rcp);
        o[6]=(__bf16)(b.z*rcp); o[7]=(__bf16)(b.w*rcp);
        *(v8bf*)(bf_out + (size_t)row * DD + lane * 8) = o;
    }
}

// ---------------------------------------------------------------------------
// Kernel 2: WMMA cosine-sim + streaming per-chunk top-5.
// grid = (NQG, NCHUNK); block = 256 (8 waves). Each WG owns 32 queries
// (2 A tiles, bf16, pre-normalized); each wave streams an interleaved subset
// of the chunk's key tiles, sharing one B fragment across both A tiles.
// PRE=true : B from pre-normalized bf16 key cache (2 b128 loads / frag);
//            A fragments hoisted into registers (loop-invariant, 128 VGPRs).
// PRE=false: B from f32 keys, converted + rcp_kn-scaled on the fly; A re-read
//            from LDS each step (keeps register pressure down). Fallback.
// ---------------------------------------------------------------------------
template<bool PRE>
__global__ __launch_bounds__(256) void sim_topk_kernel(
    const float* __restrict__ query, const float* __restrict__ keys,
    const __bf16* __restrict__ kbf,
    const float* __restrict__ rcp_qn, const float* __restrict__ rcp_kn,
    float* __restrict__ pv, int* __restrict__ pi)
{
    __shared__ __bf16 lA[MWG][DD];          // 16 KB  normalized query tiles
    __shared__ float  lKn[CHUNK];           // 16 KB  (fallback path only)
    __shared__ float  stage[8][MWG][16];    // 16 KB  per-wave C staging
    __shared__ float  candV[8][MWG][TOPK];  // 5 KB
    __shared__ int    candI[8][MWG][TOPK];  // 5 KB

    const int qtile = blockIdx.x;           // [0, NQG)
    const int chunk = blockIdx.y;           // [0, NCHUNK)
    const int chunk_base = chunk * CHUNK;
    const int tid  = threadIdx.x;
    const int wid  = tid >> 5;
    const int lane = tid & 31;
    const int hl   = lane >> 4;             // half-wave: 0 or 1
    const int l15  = lane & 15;
    const int aoff = hl * 16;               // A frag K-offset (contiguous k = 16h + s, matches B)
    const int boff = hl * 16;               // B frag K-offset

    // Stage A = qn (bf16) into LDS.
    for (int i = tid; i < MWG * DD; i += 256) {
        const int m = i / DD, d = i % DD;
        const int q = qtile * MWG + m;
        lA[m][d] = (__bf16)(query[(size_t)q * DD + d] * rcp_qn[q]);
    }
    if (!PRE) {
        for (int i = tid; i < CHUNK; i += 256) lKn[i] = rcp_kn[chunk_base + i];
    }
    __syncthreads();

    // PRE path: A fragments are invariant across all key tiles -> registers.
    v16bf A0[KSTEPS], A1[KSTEPS];
    if (PRE) {
        #pragma unroll
        for (int kk = 0; kk < KSTEPS; ++kk) {
            const int k0 = kk * 32;
            const v8bf x0 = *(const v8bf*)&lA[l15][k0 + aoff];
            const v8bf x1 = *(const v8bf*)&lA[l15][k0 + aoff + 8];
            const v8bf y0 = *(const v8bf*)&lA[l15 + 16][k0 + aoff];
            const v8bf y1 = *(const v8bf*)&lA[l15 + 16][k0 + aoff + 8];
            #pragma unroll
            for (int e = 0; e < 8; ++e) {
                A0[kk][e] = x0[e]; A0[kk][8 + e] = x1[e];
                A1[kk][e] = y0[e]; A1[kk][8 + e] = y1[e];
            }
        }
    }

    Top5 t0, t1; t0.init(); t1.init();      // rows lane and lane+16 (lanes<16)

    for (int it = 0; it < TPW; ++it) {      // uniform trip count: barriers OK
        const int t     = wid + it * 8;     // key tile within chunk [0,256)
        const int kbase = chunk_base + t * 16;
        const int key   = kbase + l15;      // this lane's B row (key)

        v8f c0 = {}, c1 = {};
        #pragma unroll
        for (int kk = 0; kk < KSTEPS; ++kk) {
            const int k0 = kk * 32;
            // B fragment (shared by both A tiles)
            v16bf b;
            if (PRE) {
                b = *(const v16bf*)(kbf + (size_t)key * DD + k0 + boff);
            } else {
                const float4* kp =
                    (const float4*)(keys + (size_t)key * DD + k0 + boff);
                const float4 f0 = kp[0], f1 = kp[1], f2 = kp[2], f3 = kp[3];
                b[0]=(__bf16)f0.x;  b[1]=(__bf16)f0.y;  b[2]=(__bf16)f0.z;  b[3]=(__bf16)f0.w;
                b[4]=(__bf16)f1.x;  b[5]=(__bf16)f1.y;  b[6]=(__bf16)f1.z;  b[7]=(__bf16)f1.w;
                b[8]=(__bf16)f2.x;  b[9]=(__bf16)f2.y;  b[10]=(__bf16)f2.z; b[11]=(__bf16)f2.w;
                b[12]=(__bf16)f3.x; b[13]=(__bf16)f3.y; b[14]=(__bf16)f3.z; b[15]=(__bf16)f3.w;
            }
            v16bf a0, a1;
            if (PRE) {
                a0 = A0[kk];
                a1 = A1[kk];
            } else {
                const v8bf x0 = *(const v8bf*)&lA[l15][k0 + aoff];
                const v8bf x1 = *(const v8bf*)&lA[l15][k0 + aoff + 8];
                const v8bf y0 = *(const v8bf*)&lA[l15 + 16][k0 + aoff];
                const v8bf y1 = *(const v8bf*)&lA[l15 + 16][k0 + aoff + 8];
                #pragma unroll
                for (int e = 0; e < 8; ++e) {
                    a0[e] = x0[e]; a0[8 + e] = x1[e];
                    a1[e] = y0[e]; a1[8 + e] = y1[e];
                }
            }
            c0 = __builtin_amdgcn_wmma_f32_16x16x32_bf16(
                     false, a0, false, b, (short)0, c0, false, false);
            c1 = __builtin_amdgcn_wmma_f32_16x16x32_bf16(
                     false, a1, false, b, (short)0, c1, false, false);
        }

        // Stage scaled sims (column N = l15 for all accum rows).
        const float knr = PRE ? 1.0f : lKn[t * 16 + l15];
        #pragma unroll
        for (int r = 0; r < 8; ++r) {
            stage[wid][r + hl * 8][l15]      = PRE ? c0[r] : c0[r] * knr;
            stage[wid][16 + r + hl * 8][l15] = PRE ? c1[r] : c1[r] * knr;
        }
        __syncthreads();

        // Row-owner lanes scan 16 candidates per row; tile-max threshold.
        if (lane < 16) {
            const float* r0 = &stage[wid][lane][0];
            const float* r1 = &stage[wid][lane + 16][0];
            float mx0 = r0[0], mx1 = r1[0];
            #pragma unroll
            for (int j = 1; j < 16; ++j) {
                mx0 = fmaxf(mx0, r0[j]);
                mx1 = fmaxf(mx1, r1[j]);
            }
            if (mx0 > t0.v4) {
                #pragma unroll
                for (int j = 0; j < 16; ++j) t0.insert(r0[j], kbase + j);
            }
            if (mx1 > t1.v4) {
                #pragma unroll
                for (int j = 0; j < 16; ++j) t1.insert(r1[j], kbase + j);
            }
        }
        __syncthreads();
    }

    // Merge 8 waves' top-5s -> per-(query,chunk) top-5 in workspace.
    if (lane < 16) {
        candV[wid][lane][0]=t0.v0; candV[wid][lane][1]=t0.v1; candV[wid][lane][2]=t0.v2;
        candV[wid][lane][3]=t0.v3; candV[wid][lane][4]=t0.v4;
        candI[wid][lane][0]=t0.i0; candI[wid][lane][1]=t0.i1; candI[wid][lane][2]=t0.i2;
        candI[wid][lane][3]=t0.i3; candI[wid][lane][4]=t0.i4;
        candV[wid][lane+16][0]=t1.v0; candV[wid][lane+16][1]=t1.v1; candV[wid][lane+16][2]=t1.v2;
        candV[wid][lane+16][3]=t1.v3; candV[wid][lane+16][4]=t1.v4;
        candI[wid][lane+16][0]=t1.i0; candI[wid][lane+16][1]=t1.i1; candI[wid][lane+16][2]=t1.i2;
        candI[wid][lane+16][3]=t1.i3; candI[wid][lane+16][4]=t1.i4;
    }
    __syncthreads();
    if (wid < 2 && lane < 16) {
        const int row = wid * 16 + lane;     // [0, 32)
        Top5 m; m.init();
        for (int w = 0; w < 8; ++w)
            for (int j = 0; j < TOPK; ++j)
                m.insert(candV[w][row][j], candI[w][row][j]);
        const int q    = qtile * MWG + row;
        const int base = (q * NCHUNK + chunk) * TOPK;
        pv[base+0]=m.v0; pv[base+1]=m.v1; pv[base+2]=m.v2; pv[base+3]=m.v3; pv[base+4]=m.v4;
        pi[base+0]=m.i0; pi[base+1]=m.i1; pi[base+2]=m.i2; pi[base+3]=m.i3; pi[base+4]=m.i4;
    }
}

// ---------------------------------------------------------------------------
// Kernel 3: merge chunk candidates, gather-mean values/errors, scatter ages.
// One wave per query; 8 waves per block.
// ---------------------------------------------------------------------------
__global__ __launch_bounds__(256) void finalize_kernel(
    const float* __restrict__ pv, const int* __restrict__ pi,
    const float* __restrict__ values, const float* __restrict__ errors,
    const float* __restrict__ ages,
    float* __restrict__ out_contents, float* __restrict__ out_err,
    float* __restrict__ out_sim, float* __restrict__ out_ages)
{
    __shared__ int   sIdx[8][TOPK];
    __shared__ float sBest[8];

    const int wid  = threadIdx.x >> 5;
    const int lane = threadIdx.x & 31;
    const int q    = blockIdx.x * 8 + wid;

    if (lane == 0) {
        Top5 m; m.init();
        const int base = q * NCHUNK * TOPK;
        for (int c = 0; c < NCHUNK * TOPK; ++c)
            m.insert(pv[base + c], pi[base + c]);
        sIdx[wid][0]=m.i0; sIdx[wid][1]=m.i1; sIdx[wid][2]=m.i2;
        sIdx[wid][3]=m.i3; sIdx[wid][4]=m.i4;
        sBest[wid]=m.v0;
    }
    __syncthreads();

    // contents = mean over 5 gathered value rows; lane owns 8 dims.
    float4 a0 = {0,0,0,0}, a1 = {0,0,0,0};
    #pragma unroll
    for (int j = 0; j < TOPK; ++j) {
        const float4* vp =
            (const float4*)(values + (size_t)sIdx[wid][j] * VV + lane * 8);
        const float4 x = vp[0], y = vp[1];
        a0.x += x.x; a0.y += x.y; a0.z += x.z; a0.w += x.w;
        a1.x += y.x; a1.y += y.y; a1.z += y.z; a1.w += y.w;
    }
    const float inv = 0.2f;
    a0.x*=inv; a0.y*=inv; a0.z*=inv; a0.w*=inv;
    a1.x*=inv; a1.y*=inv; a1.z*=inv; a1.w*=inv;
    float4* op = (float4*)(out_contents + (size_t)q * VV + lane * 8);
    op[0] = a0; op[1] = a1;

    if (lane == 0) {
        float es = 0.f;
        for (int j = 0; j < TOPK; ++j) es += errors[sIdx[wid][j]];
        out_err[q] = es * inv;
        out_sim[q] = sBest[wid];
    }
    if (lane < TOPK) {                       // races write identical values
        const int ii = sIdx[wid][lane];
        out_ages[ii] = ages[ii] + 1.0f;
    }
}

// ---------------------------------------------------------------------------
extern "C" void kernel_launch(void* const* d_in, const int* in_sizes, int n_in,
                              void* d_out, int out_size, void* d_ws, size_t ws_size,
                              hipStream_t stream)
{
    const float* query  = (const float*)d_in[0];
    const float* keys   = (const float*)d_in[1];
    const float* values = (const float*)d_in[2];
    const float* errors = (const float*)d_in[3];
    const float* ages   = (const float*)d_in[4];
    // d_in[5] = topk (==5, hardcoded per reference setup)

    float* out          = (float*)d_out;
    float* out_contents = out;                       // 1024*256
    float* out_err      = out_contents + BQ * VV;    // 1024
    float* out_sim      = out_err + BQ;              // 1024
    float* out_ages     = out_sim + BQ;              // 65536

    const size_t kbf_bytes   = (size_t)SK * DD * sizeof(__bf16);   // 32 MB
    const size_t small_bytes = (size_t)(SK + BQ + 2 * BQ * NCHUNK * TOPK) * 4;
    const bool   pre         = ws_size >= kbf_bytes + small_bytes;

    char* wsp = (char*)d_ws;
    __bf16* kbf = nullptr;
    if (pre) { kbf = (__bf16*)wsp; wsp += kbf_bytes; }
    float* rcp_kn = (float*)wsp;                     // 65536
    float* rcp_qn = rcp_kn + SK;                     // 1024
    float* pv     = rcp_qn + BQ;                     // 1024*16*5
    int*   pi     = (int*)(pv + BQ * NCHUNK * TOPK); // 1024*16*5

    rownorm_kernel<<<SK / 8, 256, 0, stream>>>(keys, SK, rcp_kn, kbf,
                                               ages, out_ages);
    rownorm_kernel<<<BQ / 8, 256, 0, stream>>>(query, BQ, rcp_qn, nullptr,
                                               nullptr, nullptr);
    if (pre) {
        sim_topk_kernel<true><<<dim3(NQG, NCHUNK), 256, 0, stream>>>(
            query, keys, kbf, rcp_qn, rcp_kn, pv, pi);
    } else {
        sim_topk_kernel<false><<<dim3(NQG, NCHUNK), 256, 0, stream>>>(
            query, keys, kbf, rcp_qn, rcp_kn, pv, pi);
    }
    finalize_kernel<<<BQ / 8, 256, 0, stream>>>(
        pv, pi, values, errors, ages, out_contents, out_err, out_sim, out_ages);
}
